// SingleHeadAttention_21646635172297
// MI455X (gfx1250) — matrix operand order repaired and checked by reading, not compiled, not run on hardware
//
#include <hip/hip_runtime.h>
#include <hip/hip_bf16.h>

// ---------------------------------------------------------------------------
// Types
// ---------------------------------------------------------------------------
typedef __attribute__((ext_vector_type(16))) __bf16 v16bf;
typedef __attribute__((ext_vector_type(8)))  __bf16 v8bf;
typedef __attribute__((ext_vector_type(8)))  float  v8f;

#define BATCH 4
#define SEQ   2048
#define DMODEL 1024
#define MSROWS (BATCH * SEQ)   // 8192 merged batch*seq rows for projections

// CDNA5 async global->LDS copy (16B per lane), tracked by ASYNCcnt.
// LDS operand is the low 32 bits of the generic shared-memory pointer
// (flat LDS aperture keeps the byte offset in addr[31:0]).
#define ASYNC_B128(ldsp, gp)                                                  \
    asm volatile("global_load_async_to_lds_b128 %0, %1, off" ::"v"(          \
                     (unsigned)(size_t)(ldsp)),                               \
                 "v"((unsigned long long)(size_t)(gp))                        \
                 : "memory")

static __device__ __forceinline__ __bf16 f2bf(float f) {
    unsigned u = __builtin_bit_cast(unsigned, f);
    unsigned r = (u + 0x7FFFu + ((u >> 16) & 1u)) >> 16;
    unsigned short s = (unsigned short)r;
    return __builtin_bit_cast(__bf16, s);
}

// ---------------------------------------------------------------------------
// fp32 -> bf16 conversion
// ---------------------------------------------------------------------------
__global__ void cvt_f32_bf16(const float* __restrict__ in,
                             __bf16* __restrict__ out, long n) {
    long i = (long)blockIdx.x * blockDim.x + threadIdx.x;
    long stride = (long)gridDim.x * blockDim.x;
    for (; i < n; i += stride) out[i] = f2bf(in[i]);
}

// ---------------------------------------------------------------------------
// GEMM  C = alpha * A * B^T   (A: MxK bf16 row-major, B: NxK bf16 row-major)
// 256 threads (8 waves), tile 128x128, K staged 64-deep, double-buffered LDS
// filled with global_load_async_to_lds_b128 (ASYNCcnt-synchronized).
// Dims assumed multiples of 128/64 (true for this problem).
// CAUSAL: skip tiles entirely above the diagonal (scores GEMM).
// ---------------------------------------------------------------------------
template <bool OUT_BF16, bool CAUSAL>
__global__ __launch_bounds__(256) void gemm_abt(
    const __bf16* __restrict__ A, const __bf16* __restrict__ Bw,
    void* __restrict__ Cout, int K, int lda, int ldb, int ldc,
    long sA, long sB, long sC, float alpha) {
    const int tn = blockIdx.x, tm = blockIdx.y, bz = blockIdx.z;
    if (CAUSAL && tn * 128 > tm * 128 + 127) return;  // fully masked tile

    A  += (size_t)bz * sA;
    Bw += (size_t)bz * sB;

    __shared__ __bf16 lA[2][128][64];
    __shared__ __bf16 lB[2][128][64];

    const int t    = threadIdx.x;
    const int lrow = t >> 1;   // staging row 0..127
    const int lq   = t & 1;    // which 32-element half of the 64-deep row
    const int lane = t & 31;
    const int w    = t >> 5;
    const int wm   = w & 3;    // 4 waves along M, 32 rows each
    const int wn   = w >> 2;   // 2 waves along N, 64 cols each
    const int lm   = lane & 15;
    const int lh   = lane >> 4;

    const v8f vz = {0.f, 0.f, 0.f, 0.f, 0.f, 0.f, 0.f, 0.f};
    v8f acc[2][4];
    for (int i = 0; i < 2; ++i)
        for (int j = 0; j < 4; ++j) acc[i][j] = vz;

    const __bf16* aSrc = A  + (size_t)(tm * 128 + lrow) * lda + lq * 32;
    const __bf16* bSrc = Bw + (size_t)(tn * 128 + lrow) * ldb + lq * 32;

    auto issue_tile = [&](int buf, int kt) {
        const __bf16* ga = aSrc + kt;
        const __bf16* gb = bSrc + kt;
        __bf16* la = &lA[buf][lrow][lq * 32];
        __bf16* lb = &lB[buf][lrow][lq * 32];
#pragma unroll
        for (int c = 0; c < 4; ++c) {
            ASYNC_B128(la + c * 8, ga + c * 8);
            ASYNC_B128(lb + c * 8, gb + c * 8);
        }
    };

    issue_tile(0, 0);  // prologue

    for (int kt = 0; kt < K; kt += 64) {
        const int cur = (kt >> 6) & 1;
        if (kt + 64 < K) {
            issue_tile(cur ^ 1, kt + 64);  // prefetch next stage
            asm volatile("s_wait_asynccnt 8" ::: "memory");  // cur stage done
        } else {
            asm volatile("s_wait_asynccnt 0" ::: "memory");
        }
        __syncthreads();

#pragma unroll
        for (int kc = 0; kc < 2; ++kc) {
            // A fragments: lane = row m, contiguous K at (lh*16)
            v16bf af[2];
#pragma unroll
            for (int mi = 0; mi < 2; ++mi) {
                const int m = wm * 32 + mi * 16 + lm;
                v8bf lo = *(const v8bf*)&lA[cur][m][kc * 32 + lh * 16];
                v8bf hi = *(const v8bf*)&lA[cur][m][kc * 32 + lh * 16 + 8];
                af[mi] = __builtin_shufflevector(lo, hi, 0, 1, 2, 3, 4, 5, 6,
                                                 7, 8, 9, 10, 11, 12, 13, 14,
                                                 15);
            }
            // B fragments: lane = col n, contiguous K at (lh*16)
            v16bf bfb[4];
#pragma unroll
            for (int ni = 0; ni < 4; ++ni) {
                const int n = wn * 64 + ni * 16 + lm;
                bfb[ni] = *(const v16bf*)&lB[cur][n][kc * 32 + lh * 16];
            }
#pragma unroll
            for (int mi = 0; mi < 2; ++mi)
#pragma unroll
                for (int ni = 0; ni < 4; ++ni)
                    acc[mi][ni] = __builtin_amdgcn_wmma_f32_16x16x32_bf16(
                        false, af[mi], false, bfb[ni], (short)0, acc[mi][ni],
                        false, false);
        }
        __syncthreads();
    }

    // Store C: VGPR r -> row (lh*8 + r), col lm within each 16x16 tile
#pragma unroll
    for (int mi = 0; mi < 2; ++mi)
#pragma unroll
        for (int ni = 0; ni < 4; ++ni) {
            const int mbase = tm * 128 + wm * 32 + mi * 16 + lh * 8;
            const int n     = tn * 128 + wn * 64 + ni * 16 + lm;
#pragma unroll
            for (int r = 0; r < 8; ++r) {
                const float v = acc[mi][ni][r] * alpha;
                const size_t idx =
                    (size_t)bz * sC + (size_t)(mbase + r) * ldc + n;
                if (OUT_BF16)
                    ((__bf16*)Cout)[idx] = f2bf(v);
                else
                    ((float*)Cout)[idx] = v;
            }
        }
}

// ---------------------------------------------------------------------------
// GEMM  C = A * B   (A: MxK bf16 row-major, B: KxN bf16 row-major)
// A staged with async copies (double-buffered); B staged TRANSPOSED into LDS
// through registers (a DMA cannot transpose) so B-fragments are contiguous-K.
// CAUSAL_K: A (= softmax probabilities) is lower-triangular, so the
// K-reduction for output tile tm only needs kt < (tm+1)*128 (multiple of the
// 64-deep stage) — halves the P*V work.
// ---------------------------------------------------------------------------
template <bool CAUSAL_K>
__global__ __launch_bounds__(256) void gemm_pv(
    const __bf16* __restrict__ A, const __bf16* __restrict__ V,
    __bf16* __restrict__ Cout, int K, int lda, int ldb, int ldc, long sA,
    long sB, long sC) {
    const int tn = blockIdx.x, tm = blockIdx.y, bz = blockIdx.z;
    A += (size_t)bz * sA;
    V += (size_t)bz * sB;

    const int Keff = CAUSAL_K ? min(K, (tm + 1) * 128) : K;

    __shared__ __bf16 lA[2][128][64];
    __shared__ __bf16 lBT[2][128][64];  // [n][k] (transposed V tile)

    const int t    = threadIdx.x;
    const int lrow = t >> 1;
    const int lq   = t & 1;
    const int lane = t & 31;
    const int w    = t >> 5;
    const int wm   = w & 3;
    const int wn   = w >> 2;
    const int lm   = lane & 15;
    const int lh   = lane >> 4;

    const v8f vz = {0.f, 0.f, 0.f, 0.f, 0.f, 0.f, 0.f, 0.f};
    v8f acc[2][4];
    for (int i = 0; i < 2; ++i)
        for (int j = 0; j < 4; ++j) acc[i][j] = vz;

    const __bf16* aSrc = A + (size_t)(tm * 128 + lrow) * lda + lq * 32;

    auto issue_tileA = [&](int buf, int kt) {
        const __bf16* ga = aSrc + kt;
        __bf16* la = &lA[buf][lrow][lq * 32];
#pragma unroll
        for (int c = 0; c < 4; ++c) ASYNC_B128(la + c * 8, ga + c * 8);
    };

    issue_tileA(0, 0);

    for (int kt = 0; kt < Keff; kt += 64) {
        const int cur = (kt >> 6) & 1;

        // B tile (64 x 128) gathered through registers, 2 tasks per thread
        union {
            uint4  u[2];
            __bf16 h[16];
        } tb[2];
#pragma unroll
        for (int j = 0; j < 2; ++j) {
            const int id  = t + 256 * j;
            const int kk  = id & 63;
            const int nch = id >> 6;
            const uint4* gv =
                (const uint4*)(V + (size_t)(kt + kk) * ldb + tn * 128 +
                               nch * 16);
            tb[j].u[0] = gv[0];
            tb[j].u[1] = gv[1];
        }

        if (kt + 64 < Keff) {
            issue_tileA(cur ^ 1, kt + 64);
            asm volatile("s_wait_asynccnt 4" ::: "memory");
        } else {
            asm volatile("s_wait_asynccnt 0" ::: "memory");
        }

        // transpose-store B into current buffer (prior readers done: the
        // last compute on this buffer finished before the previous barrier)
#pragma unroll
        for (int j = 0; j < 2; ++j) {
            const int id  = t + 256 * j;
            const int kk  = id & 63;
            const int nch = id >> 6;
#pragma unroll
            for (int e = 0; e < 16; ++e)
                lBT[cur][nch * 16 + e][kk] = tb[j].h[e];
        }
        __syncthreads();

#pragma unroll
        for (int kc = 0; kc < 2; ++kc) {
            v16bf af[2];
#pragma unroll
            for (int mi = 0; mi < 2; ++mi) {
                const int m = wm * 32 + mi * 16 + lm;
                v8bf lo = *(const v8bf*)&lA[cur][m][kc * 32 + lh * 16];
                v8bf hi = *(const v8bf*)&lA[cur][m][kc * 32 + lh * 16 + 8];
                af[mi] = __builtin_shufflevector(lo, hi, 0, 1, 2, 3, 4, 5, 6,
                                                 7, 8, 9, 10, 11, 12, 13, 14,
                                                 15);
            }
            v16bf bfb[4];
#pragma unroll
            for (int ni = 0; ni < 4; ++ni) {
                const int n = wn * 64 + ni * 16 + lm;
                bfb[ni] = *(const v16bf*)&lBT[cur][n][kc * 32 + lh * 16];
            }
#pragma unroll
            for (int mi = 0; mi < 2; ++mi)
#pragma unroll
                for (int ni = 0; ni < 4; ++ni)
                    acc[mi][ni] = __builtin_amdgcn_wmma_f32_16x16x32_bf16(
                        false, af[mi], false, bfb[ni], (short)0, acc[mi][ni],
                        false, false);
        }
        __syncthreads();
    }

#pragma unroll
    for (int mi = 0; mi < 2; ++mi)
#pragma unroll
        for (int ni = 0; ni < 4; ++ni) {
            const int mbase = tm * 128 + wm * 32 + mi * 16 + lh * 8;
            const int n     = tn * 128 + wn * 64 + ni * 16 + lm;
#pragma unroll
            for (int r = 0; r < 8; ++r) {
                const size_t idx =
                    (size_t)bz * sC + (size_t)(mbase + r) * ldc + n;
                Cout[idx] = f2bf(acc[mi][ni][r]);
            }
        }
}

// ---------------------------------------------------------------------------
// Causal row softmax: fp32 scores row [0..i] -> bf16 probabilities (zeros
// beyond the diagonal so P*V can be a dense GEMM).
// ---------------------------------------------------------------------------
__global__ __launch_bounds__(256) void softmax_causal(
    const float* __restrict__ S, __bf16* __restrict__ P, int seqlen) {
    const int b = blockIdx.y;
    const int i = blockIdx.x;
    const float* row  = S + ((size_t)b * seqlen + i) * seqlen;
    __bf16*      prow = P + ((size_t)b * seqlen + i) * seqlen;
    __shared__ float red[256];
    const int tid = threadIdx.x;

    float m = -3.402823466e38f;
    for (int j = tid; j <= i; j += 256) m = fmaxf(m, row[j]);
    red[tid] = m;
    __syncthreads();
    for (int s = 128; s > 0; s >>= 1) {
        if (tid < s) red[tid] = fmaxf(red[tid], red[tid + s]);
        __syncthreads();
    }
    const float rmax = red[0];
    __syncthreads();

    float sum = 0.f;
    for (int j = tid; j <= i; j += 256) sum += __expf(row[j] - rmax);
    red[tid] = sum;
    __syncthreads();
    for (int s = 128; s > 0; s >>= 1) {
        if (tid < s) red[tid] += red[tid + s];
        __syncthreads();
    }
    const float inv = 1.0f / red[0];

    for (int j = tid; j < seqlen; j += 256) {
        const float v = (j <= i) ? __expf(row[j] - rmax) * inv : 0.0f;
        prow[j] = f2bf(v);
    }
}

// ---------------------------------------------------------------------------
// Host-side orchestration (graph-capture safe: launches only)
// ---------------------------------------------------------------------------
extern "C" void kernel_launch(void* const* d_in, const int* in_sizes, int n_in,
                              void* d_out, int out_size, void* d_ws,
                              size_t ws_size, hipStream_t stream) {
    (void)in_sizes; (void)n_in; (void)out_size; (void)ws_size;

    const float* x  = (const float*)d_in[0];
    const float* Wq = (const float*)d_in[1];
    const float* Wk = (const float*)d_in[2];
    const float* Wv = (const float*)d_in[3];
    const float* Wo = (const float*)d_in[4];

    char*  base = (char*)d_ws;
    size_t off  = 0;
    auto take = [&](size_t bytes) -> void* {
        void* p = base + off;
        off += (bytes + 255) & ~(size_t)255;
        return p;
    };

    __bf16* xb  = (__bf16*)take((size_t)MSROWS * DMODEL * 2);
    __bf16* wqb = (__bf16*)take((size_t)DMODEL * DMODEL * 2);
    __bf16* wkb = (__bf16*)take((size_t)DMODEL * DMODEL * 2);
    __bf16* wvb = (__bf16*)take((size_t)DMODEL * DMODEL * 2);
    __bf16* wob = (__bf16*)take((size_t)DMODEL * DMODEL * 2);
    __bf16* Qb  = (__bf16*)take((size_t)MSROWS * DMODEL * 2);
    __bf16* Kb  = (__bf16*)take((size_t)MSROWS * DMODEL * 2);
    __bf16* Vb  = (__bf16*)take((size_t)MSROWS * DMODEL * 2);
    float*  sc  = (float*)take((size_t)BATCH * SEQ * SEQ * 4);
    __bf16* Pb  = (__bf16*)take((size_t)BATCH * SEQ * SEQ * 2);
    __bf16* ao  = (__bf16*)take((size_t)MSROWS * DMODEL * 2);

    // 1) fp32 -> bf16
    cvt_f32_bf16<<<2048, 256, 0, stream>>>(x, xb, (long)MSROWS * DMODEL);
    cvt_f32_bf16<<<1024, 256, 0, stream>>>(Wq, wqb, (long)DMODEL * DMODEL);
    cvt_f32_bf16<<<1024, 256, 0, stream>>>(Wk, wkb, (long)DMODEL * DMODEL);
    cvt_f32_bf16<<<1024, 256, 0, stream>>>(Wv, wvb, (long)DMODEL * DMODEL);
    cvt_f32_bf16<<<1024, 256, 0, stream>>>(Wo, wob, (long)DMODEL * DMODEL);

    // 2) Projections: Q/K/V = x @ W^T  (bf16 out)
    const dim3 gProj(DMODEL / 128, MSROWS / 128, 1);
    gemm_abt<true, false><<<gProj, 256, 0, stream>>>(
        xb, wqb, (void*)Qb, DMODEL, DMODEL, DMODEL, DMODEL, 0, 0, 0, 1.0f);
    gemm_abt<true, false><<<gProj, 256, 0, stream>>>(
        xb, wkb, (void*)Kb, DMODEL, DMODEL, DMODEL, DMODEL, 0, 0, 0, 1.0f);
    gemm_abt<true, false><<<gProj, 256, 0, stream>>>(
        xb, wvb, (void*)Vb, DMODEL, DMODEL, DMODEL, DMODEL, 0, 0, 0, 1.0f);

    // 3) scores = scale * Q @ K^T  (fp32 out, causal tile skip)
    const float scale = 1.0f / 32.0f;  // 1/sqrt(1024)
    const dim3 gScore(SEQ / 128, SEQ / 128, BATCH);
    gemm_abt<false, true><<<gScore, 256, 0, stream>>>(
        Qb, Kb, (void*)sc, DMODEL, DMODEL, DMODEL, SEQ, (long)SEQ * DMODEL,
        (long)SEQ * DMODEL, (long)SEQ * SEQ, scale);

    // 4) causal softmax -> bf16 probabilities
    softmax_causal<<<dim3(SEQ, BATCH), 256, 0, stream>>>(sc, Pb, SEQ);

    // 5) attn = P @ V  (bf16 out, K-reduction clipped at the diagonal)
    const dim3 gPV(DMODEL / 128, SEQ / 128, BATCH);
    gemm_pv<true><<<gPV, 256, 0, stream>>>(Pb, Vb, ao, SEQ, SEQ, DMODEL,
                                           DMODEL, (long)SEQ * SEQ,
                                           (long)SEQ * DMODEL,
                                           (long)SEQ * DMODEL);

    // 6) out = attn @ W_O^T  (fp32 out to d_out)
    gemm_abt<false, false><<<gProj, 256, 0, stream>>>(
        ao, wob, d_out, DMODEL, DMODEL, DMODEL, DMODEL, 0, 0, 0, 1.0f);
}
